// SelfAttention_49108656062926
// MI455X (gfx1250) — matrix operand order repaired and checked by reading, not compiled, not run on hardware
//
#include <hip/hip_runtime.h>

// MI455X (gfx1250) fused self-attention with edge-feature bias.
// wave32, WMMA f32<=f16 16x16x32 for all K>=32 contractions; fp32 stats/softmax.
// B=4 L=512 D=256 H=8 hd=32 E=5.

typedef __attribute__((ext_vector_type(16))) _Float16 v16h;
typedef __attribute__((ext_vector_type(8)))  float    v8f;
typedef __attribute__((ext_vector_type(2)))  _Float16 h2;

#define B_  4
#define L_  512
#define D_  256
#define H_  8
#define HD_ 32
#define E_  5

static __device__ __forceinline__ v8f wmma16x16x32(v16h a, v16h b, v8f c) {
  return __builtin_amdgcn_wmma_f32_16x16x32_f16(false, a, false, b, (short)0, c,
                                                false, false);
}

// ---------------------------------------------------------------------------
// Kernel 0: one-time weight prep.  Transpose the four fp32 [in][out] weights
// to f16 [out][in] so WMMA B-fragments load as contiguous 32 B per lane.
// 512 KB total -> L2-resident for all later GEMMs.
// ---------------------------------------------------------------------------
__global__ __launch_bounds__(256)
void weight_prep_kernel(const float* __restrict__ Wq, const float* __restrict__ Wk,
                        const float* __restrict__ Wv, const float* __restrict__ Wp,
                        _Float16* __restrict__ Wqt, _Float16* __restrict__ Wkt,
                        _Float16* __restrict__ Wvt, _Float16* __restrict__ Wpt) {
  const int idx = blockIdx.x * 256 + threadIdx.x;   // 4 * 65536 total
  const int m   = idx >> 16;
  const int rem = idx & 65535;
  const int c   = rem >> 8;     // out col
  const int k   = rem & 255;    // in  row
  const float* src = (m == 0) ? Wq : (m == 1) ? Wk : (m == 2) ? Wv : Wp;
  _Float16*    dst = (m == 0) ? Wqt : (m == 1) ? Wkt : (m == 2) ? Wvt : Wpt;
  dst[c * D_ + k] = (_Float16)src[k * D_ + c];
}

// ---------------------------------------------------------------------------
// Kernel 1: QKV projection.  X[2048x256] fp32 @ W + bias -> f16.
// VT=false: head-major [B][H][L][hd] (Q, K).
// VT=true : transposed [B][H][hd][L] (V) so P@V B-frags are contiguous.
// One wave per 16x16 output tile.
// ---------------------------------------------------------------------------
template <bool VT>
__global__ __launch_bounds__(32)
void proj_kernel(const float* __restrict__ X, const _Float16* __restrict__ Wt,
                 const float* __restrict__ bias, _Float16* __restrict__ out) {
  const int lane = threadIdx.x;
  const int n    = lane & 15;
  const int hi   = lane >> 4;
  const int ako  = hi ? 16 : 0;   // A-frag K striping (contiguous k = 16h + s)
  const int bko  = hi ? 16 : 0;   // B-frag K striping (contiguous k = 16h + s, same as A)
  const int rt   = blockIdx.x;    // row tile (16 rows of 2048)
  const int ct   = blockIdx.y;    // col tile (16 cols of 256)

  v8f c = {};
  const float*    xr = X + (rt * 16 + n) * D_;   // A row m == lane&15
  const _Float16* wc = Wt + (ct * 16 + n) * D_;  // B col n == lane&15, contiguous K
#pragma unroll
  for (int kb = 0; kb < 8; ++kb) {
    const int kbase = kb * 32;
    v16h a, b;
#pragma unroll
    for (int e = 0; e < 8; ++e) {
      a[e]     = (_Float16)xr[kbase + ako + e];
      a[e + 8] = (_Float16)xr[kbase + ako + 8 + e];
    }
#pragma unroll
    for (int e = 0; e < 16; ++e)
      b[e] = wc[kbase + bko + e];
    c = wmma16x16x32(a, b, c);
  }
  const int   col = ct * 16 + n;
  const float bb  = bias[col];
  const int   h   = col >> 5;
  const int   dd  = col & 31;
#pragma unroll
  for (int r = 0; r < 8; ++r) {
    const int row = rt * 16 + r + hi * 8;     // C row M = r + 8*(lane>=16)
    const int b0  = row >> 9;
    const int l   = row & 511;
    if (VT)
      out[((b0 * H_ + h) * HD_ + dd) * L_ + l] = (_Float16)(c[r] + bb);
    else
      out[((b0 * H_ + h) * L_ + l) * HD_ + dd] = (_Float16)(c[r] + bb);
  }
}

// ---------------------------------------------------------------------------
// Kernel 2: fused edge MLP -> per-head additive bias  ebias[B][L][L][H].
// LN folded algebraically: bias_h = rsig*(dot(e, g.We2[:,h]) - mu*colsum_h)
//                                   + (ln_b.We2[:,h] + be2_h).
// Stage A (VALU, K=5): e = relu(ef@We1+be1) -> f16 in LDS + s1/s2 stats.
// Stage B (WMMA):      [16 pairs x 256] @ Gt[256x16] (8 heads + zero pad).
// 4 waves/block, each wave owns one 16-pair tile (fixed i, 16 consecutive j).
// ---------------------------------------------------------------------------
__global__ __launch_bounds__(128)
void edge_bias_kernel(const float* __restrict__ edge, const float* __restrict__ We1,
                      const float* __restrict__ be1,  const float* __restrict__ ln_g,
                      const float* __restrict__ ln_b, const float* __restrict__ We2,
                      const float* __restrict__ be2,  float* __restrict__ ebias) {
  __shared__ float    We1_s[E_ * D_];
  __shared__ float    be1_s[D_];
  __shared__ _Float16 Gt_s[16 * D_];     // Gt[n][d] = ln_g[d]*We2[d][n], n<8; else 0
  __shared__ float    colsum_s[H_];
  __shared__ float    cconst_s[H_];
  __shared__ _Float16 Es[4 * 16 * D_];   // per-wave e-matrix, 16 pairs x 256, f16
  __shared__ float    mu_s[4][16];
  __shared__ float    rs_s[4][16];

  const int tid  = threadIdx.x;
  const int w    = tid >> 5;
  const int lane = tid & 31;
  const int p    = lane & 15;            // pair within tile
  const int hi   = lane >> 4;
  const int t    = blockIdx.x * 4 + w;   // tile id over (b, i, jt)
  const int b0   = t >> 14;
  const int i    = (t >> 5) & 511;
  const int jt   = t & 31;
  const int j    = jt * 16 + p;

  const float* ef = edge + (size_t)((b0 * L_ + i) * L_ + j) * E_;
  __builtin_prefetch(ef, 0, 0);          // global_prefetch_b8: stream-ahead

  for (int idx = tid; idx < E_ * D_; idx += 128) We1_s[idx] = We1[idx];
  for (int idx = tid; idx < D_;      idx += 128) be1_s[idx] = be1[idx];
  for (int idx = tid; idx < 16 * D_; idx += 128) {
    const int hh = idx >> 8, d = idx & 255;
    Gt_s[hh * D_ + d] =
        (hh < H_) ? (_Float16)(ln_g[d] * We2[d * H_ + hh]) : (_Float16)0.0f;
  }
  if (tid < H_) {
    float cs = 0.f, cc = 0.f;
    for (int d = 0; d < D_; ++d) {
      cs += ln_g[d] * We2[d * H_ + tid];
      cc += ln_b[d] * We2[d * H_ + tid];
    }
    colsum_s[tid] = cs;
    cconst_s[tid] = cc + be2[tid];
  }
  __syncthreads();

  // Stage A: K=5 linear + ReLU (lane owns half the d-range of pair p).
  const float f0 = ef[0], f1 = ef[1], f2 = ef[2], f3 = ef[3], f4 = ef[4];
  float s1 = 0.f, s2 = 0.f;
  const int dbase = hi * 128;
  h2* Erow2 = (h2*)&Es[(w * 16 + p) * D_ + dbase];
  for (int dd = 0; dd < 128; dd += 2) {
    float v2[2];
#pragma unroll
    for (int u = 0; u < 2; ++u) {
      const int d = dbase + dd + u;
      float v = be1_s[d];
      v = fmaf(f0, We1_s[0 * D_ + d], v);
      v = fmaf(f1, We1_s[1 * D_ + d], v);
      v = fmaf(f2, We1_s[2 * D_ + d], v);
      v = fmaf(f3, We1_s[3 * D_ + d], v);
      v = fmaf(f4, We1_s[4 * D_ + d], v);
      v = fmaxf(v, 0.f);
      s1 += v;
      s2 = fmaf(v, v, s2);
      v2[u] = v;
    }
    h2 pk;
    pk[0] = (_Float16)v2[0];
    pk[1] = (_Float16)v2[1];
    Erow2[dd >> 1] = pk;                 // one ds_store_b32 per 2 elements
  }
  s1 += __shfl_xor(s1, 16, 32);
  s2 += __shfl_xor(s2, 16, 32);
  if (lane < 16) {
    const float mu  = s1 * (1.f / 256.f);
    const float var = s2 * (1.f / 256.f) - mu * mu;
    mu_s[w][p] = mu;
    rs_s[w][p] = rsqrtf(var + 1e-5f);
  }
  __syncthreads();

  // Stage B: WMMA over K=256 (8 steps), A = e-matrix rows (pairs), B = Gt cols.
  const int ako = hi ? 16 : 0;
  const int bko = hi ? 16 : 0;
  const _Float16* Eb = &Es[(w * 16 + p) * D_];
  const _Float16* Gn = &Gt_s[p * D_];    // B column n == lane&15
  v8f c = {};
#pragma unroll
  for (int kb = 0; kb < 8; ++kb) {
    const int kbase = kb * 32;
    v16h a, b;
#pragma unroll
    for (int e = 0; e < 8; ++e) {
      a[e]     = Eb[kbase + ako + e];
      a[e + 8] = Eb[kbase + ako + 8 + e];
    }
#pragma unroll
    for (int e = 0; e < 16; ++e)
      b[e] = Gn[kbase + bko + e];
    c = wmma16x16x32(a, b, c);
  }
  const int n = lane & 15;
  if (n < H_) {
#pragma unroll
    for (int r = 0; r < 8; ++r) {
      const int   m  = r + hi * 8;       // pair index (C row)
      const float bv = rs_s[w][m] * (c[r] - mu_s[w][m] * colsum_s[n]) + cconst_s[n];
      ebias[(size_t)((b0 * L_ + i) * L_ + (jt * 16 + m)) * H_ + n] = bv;
    }
  }
}

// ---------------------------------------------------------------------------
// Kernel 3: attention.  One wave per (b, q-tile, head).
// S[16x512] scores held in LDS (32 KB of the 320 KB WGP pool).
// ---------------------------------------------------------------------------
__global__ __launch_bounds__(32)
void attention_kernel(const _Float16* __restrict__ qh, const _Float16* __restrict__ kh,
                      const _Float16* __restrict__ vt, const int* __restrict__ mask,
                      const float* __restrict__ ebias, float* __restrict__ y) {
  __shared__ float S[16 * L_];
  __shared__ float inv_s[16];

  const int bid  = blockIdx.x;
  const int h    = bid & 7;
  const int it   = (bid >> 3) & 31;
  const int b0   = bid >> 8;
  const int lane = threadIdx.x;
  const int nn   = lane & 15;
  const int hi   = lane >> 4;
  const int ako  = hi ? 16 : 0;
  const int bko  = hi ? 16 : 0;
  const float NEG_INF = -__builtin_inff();

  const _Float16* qbase = qh + (size_t)((b0 * H_ + h) * L_) * HD_;
  const _Float16* kbase = kh + (size_t)((b0 * H_ + h) * L_) * HD_;
  const _Float16* vbase = vt + (size_t)((b0 * H_ + h) * HD_) * L_;  // [hd][L]

  // Phase 1: scores.  K=32 == hd, so one WMMA per 16-key tile.
  v16h a;
  {
    const _Float16* qr = qbase + (it * 16 + nn) * HD_;
#pragma unroll
    for (int e = 0; e < 8; ++e) { a[e] = qr[ako + e]; a[e + 8] = qr[ako + 8 + e]; }
  }
  const float scale = 0.17677669529663687f;  // 1/sqrt(32)
  for (int kt = 0; kt < 32; ++kt) {
    const _Float16* kr = kbase + (kt * 16 + nn) * HD_ + bko;  // B[k][n] = k[n][k]
    if (kt + 1 < 32)
      __builtin_prefetch(kr + 16 * HD_, 0, 0);
    v16h bf;
#pragma unroll
    for (int e = 0; e < 16; ++e) bf[e] = kr[e];
    v8f c = {};
    c = wmma16x16x32(a, bf, c);
#pragma unroll
    for (int r = 0; r < 8; ++r)
      S[(r + hi * 8) * L_ + kt * 16 + nn] = c[r] * scale;
  }
  __syncthreads();

  // Phase 2: add precomputed edge bias, apply mask (0 -> -inf).
  const int*   mrow = mask + (b0 * L_ + it * 16) * L_;
  const float* erow = ebias + (size_t)(b0 * L_ + it * 16) * L_ * H_ + h;
  for (int idx = lane; idx < 16 * L_; idx += 32) {
    const int m = idx >> 9;
    const int j = idx & 511;
    float v = S[idx] + erow[(size_t)(m * L_ + j) * H_];
    S[idx]  = (mrow[m * L_ + j] != 0) ? v : NEG_INF;
  }
  __syncthreads();

  // Phase 3: softmax over keys (2 lanes per row, shfl_xor combine).
  {
    const int row  = lane >> 1;
    const int half = lane & 1;
    const int j0   = half * 256;
    float mx = NEG_INF;
    for (int j = 0; j < 256; ++j) mx = fmaxf(mx, S[row * L_ + j0 + j]);
    mx = fmaxf(mx, __shfl_xor(mx, 1, 32));
    float sum = 0.f;
    for (int j = 0; j < 256; ++j) {
      const float p = __expf(S[row * L_ + j0 + j] - mx);
      S[row * L_ + j0 + j] = p;
      sum += p;
    }
    sum += __shfl_xor(sum, 1, 32);
    if (half == 0) inv_s[row] = 1.f / sum;
  }
  __syncthreads();

  // Phase 4: y = P @ V (normalization folded into epilogue).
  // V is transposed [hd][L], so B-frag K-range is contiguous per lane.
  v8f c0 = {}, c1 = {};
  for (int kt = 0; kt < 16; ++kt) {
    v16h af;
    const float* srow = &S[nn * L_ + kt * 32];  // A row = query nn
#pragma unroll
    for (int e = 0; e < 8; ++e) {
      af[e]     = (_Float16)srow[ako + e];
      af[e + 8] = (_Float16)srow[ako + 8 + e];
    }
    v16h bf0, bf1;
    const _Float16* vr0 = vbase + nn * L_ + kt * 32 + bko;
    const _Float16* vr1 = vbase + (16 + nn) * L_ + kt * 32 + bko;
#pragma unroll
    for (int e = 0; e < 16; ++e) {
      bf0[e] = vr0[e];
      bf1[e] = vr1[e];
    }
    c0 = wmma16x16x32(af, bf0, c0);
    c1 = wmma16x16x32(af, bf1, c1);
  }
#pragma unroll
  for (int r = 0; r < 8; ++r) {
    const int   m    = r + hi * 8;
    const float isum = inv_s[m];
    const int   l    = it * 16 + m;
    float* yr = y + (size_t)((b0 * L_ + l) * D_) + h * HD_;
    yr[nn]      = c0[r] * isum;
    yr[16 + nn] = c1[r] * isum;
  }
}

// ---------------------------------------------------------------------------
// Kernel 4: output projection  y[2048x256] @ Wp + bp -> out fp32 (row-major).
// ---------------------------------------------------------------------------
__global__ __launch_bounds__(32)
void out_proj_kernel(const float* __restrict__ Y, const _Float16* __restrict__ Wpt,
                     const float* __restrict__ bp, float* __restrict__ out) {
  const int lane = threadIdx.x;
  const int n    = lane & 15;
  const int hi   = lane >> 4;
  const int ako  = hi ? 16 : 0;
  const int bko  = hi ? 16 : 0;
  const int rt   = blockIdx.x;
  const int ct   = blockIdx.y;

  v8f c = {};
  const float*    yr = Y + (rt * 16 + n) * D_;
  const _Float16* wc = Wpt + (ct * 16 + n) * D_;
#pragma unroll
  for (int kb = 0; kb < 8; ++kb) {
    const int kbase = kb * 32;
    v16h a, b;
#pragma unroll
    for (int e = 0; e < 8; ++e) {
      a[e]     = (_Float16)yr[kbase + ako + e];
      a[e + 8] = (_Float16)yr[kbase + ako + 8 + e];
    }
#pragma unroll
    for (int e = 0; e < 16; ++e)
      b[e] = wc[kbase + bko + e];
    c = wmma16x16x32(a, b, c);
  }
  const float bb = bp[ct * 16 + n];
#pragma unroll
  for (int r = 0; r < 8; ++r)
    out[(rt * 16 + r + hi * 8) * D_ + ct * 16 + n] = c[r] + bb;
}

// ---------------------------------------------------------------------------
extern "C" void kernel_launch(void* const* d_in, const int* in_sizes, int n_in,
                              void* d_out, int out_size, void* d_ws, size_t ws_size,
                              hipStream_t stream) {
  (void)in_sizes; (void)n_in; (void)out_size; (void)ws_size;
  const float* key   = (const float*)d_in[0];
  const float* value = (const float*)d_in[1];
  const float* query = (const float*)d_in[2];
  const int*   masks = (const int*)d_in[3];
  const float* edge  = (const float*)d_in[4];
  const float* Wk = (const float*)d_in[5];   const float* bk = (const float*)d_in[6];
  const float* Wq = (const float*)d_in[7];   const float* bq = (const float*)d_in[8];
  const float* Wv = (const float*)d_in[9];   const float* bv = (const float*)d_in[10];
  const float* Wp = (const float*)d_in[11];  const float* bp = (const float*)d_in[12];
  const float* We1 = (const float*)d_in[13]; const float* be1 = (const float*)d_in[14];
  const float* lng = (const float*)d_in[15]; const float* lnb = (const float*)d_in[16];
  const float* We2 = (const float*)d_in[17]; const float* be2 = (const float*)d_in[18];
  float* out = (float*)d_out;

  // Workspace carve-up (~37.5 MB):
  //   qh, kh  f16 [B][H][L][hd];  vt f16 [B][H][hd][L];
  //   Wqt/Wkt/Wvt/Wpt f16 [out][in];  ebias fp32 [B][L][L][H];  y fp32.
  const size_t qkv_elems = (size_t)B_ * H_ * L_ * HD_;       // 524288
  const size_t w_elems   = (size_t)D_ * D_;                  // 65536
  char* ws = (char*)d_ws;
  _Float16* qh  = (_Float16*)ws;
  _Float16* kh  = qh + qkv_elems;
  _Float16* vt  = kh + qkv_elems;
  _Float16* wqt = vt + qkv_elems;
  _Float16* wkt = wqt + w_elems;
  _Float16* wvt = wkt + w_elems;
  _Float16* wpt = wvt + w_elems;
  float* ebias = (float*)(ws + (3 * qkv_elems + 4 * w_elems) * sizeof(_Float16));
  float* yws   = ebias + (size_t)B_ * L_ * L_ * H_;          // 8M floats

  weight_prep_kernel<<<1024, 256, 0, stream>>>(Wq, Wk, Wv, Wp, wqt, wkt, wvt, wpt);

  dim3 gproj(128, 16);
  proj_kernel<false><<<gproj, 32, 0, stream>>>(query, wqt, bq, qh);
  proj_kernel<false><<<gproj, 32, 0, stream>>>(key,   wkt, bk, kh);
  proj_kernel<true ><<<gproj, 32, 0, stream>>>(value, wvt, bv, vt);

  // 4*512*32 = 65536 16-pair tiles, 4 waves (tiles) per block.
  edge_bias_kernel<<<16384, 128, 0, stream>>>(edge, We1, be1, lng, lnb, We2, be2,
                                              ebias);

  // B * L/16 * H = 1024 single-wave workgroups.
  attention_kernel<<<1024, 32, 0, stream>>>(qh, kh, vt, masks, ebias, yws);

  out_proj_kernel<<<gproj, 32, 0, stream>>>(yws, wpt, bp, out);
}
